// NetworkRNN_46059229282644
// MI455X (gfx1250) — matrix operand order repaired and checked by reading, not compiled, not run on hardware
//
#include <hip/hip_runtime.h>

// ---------------- problem constants (match reference) ----------------
#define NP 8        // populations
#define NN 2048     // units per population
#define NE 16       // recurrent edges
#define NB 4        // batch
#define NT 128      // timesteps
#define NMAT (NE + 1)       // 16 edge weight matrices + W_in
#define NSLOT (NP + 1)      // 8 populations + input population
#define NJT (NN / 16)       // 128 j-tiles of 16
#define NKC (NN / 32)       // 64  k-chunks of 32
#define DT_F 0.1f
#define TAU_S_F 5.0f
#define TAU_M_F 10.0f
#define TWO_PI_F 6.28318530717958647692f

typedef __attribute__((ext_vector_type(16))) __bf16 v16bf;
typedef __attribute__((ext_vector_type(8)))  float  v8f;

// ---------------- ws layout (bytes) ----------------
#define WS_A_OFF   ((size_t)0)
#define WS_A_SZ    ((size_t)NMAT * NJT * NKC * 1024)          // 142,606,336
#define WS_B_OFF   (WS_A_OFF + WS_A_SZ)
#define WS_B_SZ    ((size_t)NSLOT * NKC * 1024)               // 589,824
#define WS_V_OFF   (WS_B_OFF + WS_B_SZ)
#define WS_V_SZ    ((size_t)NP * NB * NN * 4)                 // 262,144
#define WS_S_OFF   (WS_V_OFF + WS_V_SZ)
#define WS_S_SZ    ((size_t)NE * NB * NN * 4)                 // 524,288
#define WS_SIN_OFF (WS_S_OFF + WS_S_SZ)
#define WS_SIN_SZ  ((size_t)NB * NN * 4)                      // 32,768
#define WS_ERR_OFF (WS_SIN_OFF + WS_SIN_SZ)
#define WS_ERR_SZ  ((size_t)NT * 256 * 4)                     // 131,072

__device__ __forceinline__ unsigned short f2bf(float f) {
    unsigned int u = __float_as_uint(f);
    unsigned int r = u + 0x7FFFu + ((u >> 16) & 1u);   // round-to-nearest-even
    return (unsigned short)(r >> 16);
}
__device__ __forceinline__ float sigmoidf(float x) {
    return 1.0f / (1.0f + __expf(-x));
}

// ---------------------------------------------------------------------------
// Prep: W (f32, [E,N,N]) + W_in -> bf16 A-fragments of W^T in WMMA lane order.
// A-tile is 16x32 (M=j rows, K=i cols). Per ISA: lane L (hi=L>>4, m=L&15),
// vgpr v, half h holds K = 2*(v&3) + h + (v>=4 ? 16 : 0) + hi*8.
// Fragment storage: lane-contiguous 32B (16 bf16 in vgpr/half order).
// ---------------------------------------------------------------------------
__global__ void k_swizzleW(const float* __restrict__ W,
                           const float* __restrict__ W_in,
                           char* __restrict__ wsA) {
    long tid = (long)blockIdx.x * blockDim.x + threadIdx.x;
    if (tid >= (long)NMAT * NJT * NKC * 32) return;
    int lane = (int)(tid & 31);
    int kc   = (int)((tid >> 5) & (NKC - 1));
    int jt   = (int)((tid >> 11) & (NJT - 1));
    int mat  = (int)(tid >> 18);
    const float* src = (mat < NE) ? (W + (size_t)mat * NN * NN) : W_in;
    int hi = lane >> 4;
    int j  = jt * 16 + (lane & 15);
    unsigned short* dst =
        (unsigned short*)(wsA + (((size_t)mat * NJT + jt) * NKC + kc) * 1024 + (size_t)lane * 32);
#pragma unroll
    for (int q = 0; q < 16; ++q) {
        int v = q >> 1, h = q & 1;
        int k = 2 * (v & 3) + h + ((v >= 4) ? 16 : 0) + hi * 8;
        int i = kc * 32 + k;
        dst[q] = f2bf(src[(size_t)i * NN + j]);   // A[m][k] = W[i][j]  (W^T)
    }
}

// ---------------------------------------------------------------------------
// Per step: build bf16 B-fragments (pre^T: 32x16, K=i rows, N=batch cols).
// Per ISA: lane L (n=L&15, hi=L>>4), vgpr v, half h holds K = 2*(v&3) + h + (v>=4 ? 16 : 0) + hi*8.
// Fragment layout per 1KB chunk: two 512B halves (VGPRs 0-3 at lane*16,
// VGPRs 4-7 at 512 + lane*16) -> conflict-free ds_load_b128 pairs.
// slot 0..7 = sigmoid(v[pop]); slot 8 = analytic input rate at t_seq[b,t].
// ---------------------------------------------------------------------------
__global__ void k_bfrag(const float* __restrict__ vstate,
                        const float* __restrict__ t_seq,
                        char* __restrict__ wsB, int step) {
    int tid = blockIdx.x * blockDim.x + threadIdx.x;
    if (tid >= NSLOT * NKC * 32) return;
    int lane = tid & 31;
    int kc   = (tid >> 5) & (NKC - 1);
    int slot = tid >> 11;
    int hi = lane >> 4, n = lane & 15;
    char* base = wsB + ((size_t)slot * NKC + kc) * 1024;
    unsigned short* dlo = (unsigned short*)(base + (size_t)lane * 16);
    unsigned short* dhi = (unsigned short*)(base + 512 + (size_t)lane * 16);
    float vals[16];
    if (n < NB) {
        if (slot < NP) {
            const float* vp = vstate + ((size_t)slot * NB + n) * NN;
#pragma unroll
            for (int q = 0; q < 16; ++q) {
                int i = kc * 32 + hi * 8 + q + (q & 8);     // K = hi*8 + q + (q>=8 ? 8 : 0)
                vals[q] = sigmoidf(vp[i]);
            }
        } else {
            float t    = t_seq[n * NT + step];
            float base_ph = TWO_PI_F * t * 0.01f;
#pragma unroll
            for (int q = 0; q < 16; ++q) {
                int i = kc * 32 + hi * 8 + q + (q & 8);
                float ph = (float)i * (TWO_PI_F / (float)NN);
                vals[q] = 0.5f * (1.0f + __sinf(base_ph + ph));
            }
        }
    } else {
#pragma unroll
        for (int q = 0; q < 16; ++q) vals[q] = 0.0f;
    }
#pragma unroll
    for (int q = 0; q < 8; ++q) dlo[q] = f2bf(vals[q]);
#pragma unroll
    for (int q = 0; q < 8; ++q) dhi[q] = f2bf(vals[q + 8]);
}

// ---------------------------------------------------------------------------
// Per step: per (edge, j-tile) wave computes drive[:, jtile] via WMMA over
// K=2048 (64 chunks of 32), then updates the exponential synapse state.
// All 8 waves of a block share one edge -> one src slot: B fragments (64KB)
// are staged in LDS once per block, halving global/L2 traffic.
// D[m][n] = sum_k W^T[j0+m][k] * pre^T[k][n] = drive[b=n][j0+m].
// ---------------------------------------------------------------------------
__global__ void __launch_bounds__(256)
k_edge_gemm(const char* __restrict__ wsA,
            const char* __restrict__ wsB,
            float* __restrict__ sstate,
            float* __restrict__ sinstate,
            const int* __restrict__ src) {
    __shared__ uint4 ldsB[4096];               // 64KB: all B fragments of slot

    int e    = blockIdx.x;                     // 0..16 (16 == input synapse)
    int wave = threadIdx.x >> 5;               // 0..7
    int lane = threadIdx.x & 31;
    int jt   = blockIdx.y * 8 + wave;          // 0..127
    int slot = (e < NE) ? src[e] : NP;

    // Cooperative stage of the whole slot's B fragments into LDS.
    const uint4* gB = (const uint4*)(wsB + (size_t)slot * NKC * 1024);
#pragma unroll
    for (int i = 0; i < 16; ++i)
        ldsB[i * 256 + threadIdx.x] = gB[i * 256 + threadIdx.x];
    __syncthreads();

    const char* aBase = wsA + ((size_t)e * NJT + jt) * NKC * 1024 + (size_t)lane * 32;

    union BU { uint4 q[2]; v16bf v; };
    v8f c0 = {};
    v8f c1 = {};
#pragma unroll 4
    for (int kc = 0; kc < NKC; ++kc) {
        v16bf a = *(const v16bf*)(aBase + (size_t)kc * 1024);
        BU bu;
        bu.q[0] = ldsB[kc * 64 + lane];        // VGPRs 0-3 (conflict-free)
        bu.q[1] = ldsB[kc * 64 + 32 + lane];   // VGPRs 4-7 (conflict-free)
        if (kc & 1)
            c1 = __builtin_amdgcn_wmma_f32_16x16x32_bf16(
                     false, a, false, bu.v, (short)0, c1, false, false);
        else
            c0 = __builtin_amdgcn_wmma_f32_16x16x32_bf16(
                     false, a, false, bu.v, (short)0, c0, false, false);
    }

    int bcol = lane & 15, hi = lane >> 4;
    if (bcol < NB) {
        const float decay = 1.0f - DT_F / TAU_S_F;   // 0.98
        float* sp = (e < NE) ? (sstate + ((size_t)e * NB + bcol) * NN)
                             : (sinstate + (size_t)bcol * NN);
#pragma unroll
        for (int r = 0; r < 8; ++r) {
            int j = jt * 16 + r + 8 * hi;            // C layout: M = r + 8*hi
            float sv = sp[j];
            sp[j] = sv * decay + DT_F * (c0[r] + c1[r]);  // s + DT*(-s/tau + drive)
        }
    }
}

// ---------------------------------------------------------------------------
// Per step: segment-sum currents over edges, Euler step on v, emit
// sigmoid(v_new) into out[B,T,P,N], deterministic block partials of |dv|.
// ---------------------------------------------------------------------------
__global__ void k_update(float* __restrict__ vstate,
                         const float* __restrict__ sstate,
                         const float* __restrict__ sinstate,
                         const int* __restrict__ tgt,
                         float* __restrict__ out,
                         float* __restrict__ errPart, int step) {
    int idx = blockIdx.x * blockDim.x + threadIdx.x;   // < P*B*N = 65536
    int p   = idx >> 13;          // / (B*N)
    int rem = idx & 8191;
    int b   = rem >> 11;          // / N
    int j   = rem & (NN - 1);

    float v = vstate[idx];
    float I = 0.0f, g = 0.0f;
#pragma unroll
    for (int e = 0; e < NE; ++e) {
        if (tgt[e] == p) {
            float s = sstate[((size_t)e * NB + b) * NN + j];
            I += s * (1.0f - v);          // G_MAX=1, E_REV=1
            g += s;
        }
    }
    if (p == 0) {
        float s = sinstate[(size_t)b * NN + j];
        I += s * (1.0f - v);
        g += s;
    }
    float dv = (-v - g * v + I) * (1.0f / TAU_M_F);
    float vn = v + DT_F * dv;
    vstate[idx] = vn;
    out[(((size_t)b * NT + step) * NP + p) * NN + j] = sigmoidf(vn);

    __shared__ float red[256];
    red[threadIdx.x] = fabsf(dv);
    __syncthreads();
    for (int s = 128; s > 0; s >>= 1) {
        if (threadIdx.x < s) red[threadIdx.x] += red[threadIdx.x + s];
        __syncthreads();
    }
    if (threadIdx.x == 0) errPart[step * 256 + blockIdx.x] = red[0];
}

// Fixed-order final reduction of err partials -> scalar.
__global__ void k_err(const float* __restrict__ errPart, float* __restrict__ out_err) {
    __shared__ float red[256];
    float acc = 0.0f;
    for (int i = threadIdx.x; i < NT * 256; i += 256) acc += errPart[i];
    red[threadIdx.x] = acc;
    __syncthreads();
    for (int s = 128; s > 0; s >>= 1) {
        if (threadIdx.x < s) red[threadIdx.x] += red[threadIdx.x + s];
        __syncthreads();
    }
    if (threadIdx.x == 0)
        *out_err = red[0] * (DT_F / (float)(NP * NB * NN));   // mean(|dv|)*DT summed
}

// ---------------------------------------------------------------------------
extern "C" void kernel_launch(void* const* d_in, const int* in_sizes, int n_in,
                              void* d_out, int out_size, void* d_ws, size_t ws_size,
                              hipStream_t stream) {
    const float* t_seq = (const float*)d_in[0];   // [B,T]
    const float* W     = (const float*)d_in[1];   // [E,N,N]
    const float* W_in  = (const float*)d_in[2];   // [N,N]
    const float* v0    = (const float*)d_in[3];   // [P,B,N]
    const float* s0    = (const float*)d_in[4];   // [E,B,N]
    const float* s_in0 = (const float*)d_in[5];   // [B,N]
    const int*   src   = (const int*)d_in[6];     // [E]
    const int*   tgt   = (const int*)d_in[7];     // [E]

    char*  ws       = (char*)d_ws;
    char*  wsA      = ws + WS_A_OFF;
    char*  wsB      = ws + WS_B_OFF;
    float* vstate   = (float*)(ws + WS_V_OFF);
    float* sstate   = (float*)(ws + WS_S_OFF);
    float* sinstate = (float*)(ws + WS_SIN_OFF);
    float* errPart  = (float*)(ws + WS_ERR_OFF);

    // Re-initialize evolving state every call (deterministic; graph-safe D2D).
    hipMemcpyAsync(vstate,   v0,    WS_V_SZ,   hipMemcpyDeviceToDevice, stream);
    hipMemcpyAsync(sstate,   s0,    WS_S_SZ,   hipMemcpyDeviceToDevice, stream);
    hipMemcpyAsync(sinstate, s_in0, WS_SIN_SZ, hipMemcpyDeviceToDevice, stream);

    // bf16-swizzle all 17 weight matrices into WMMA A-fragment layout (~142MB,
    // fits the 192MB L2 so the time loop streams weights from L2, not HBM).
    long swTotal = (long)NMAT * NJT * NKC * 32;
    k_swizzleW<<<dim3((unsigned)((swTotal + 255) / 256)), dim3(256), 0, stream>>>(W, W_in, wsA);

    for (int t = 0; t < NT; ++t) {
        k_bfrag<<<dim3((NSLOT * NKC * 32 + 255) / 256), dim3(256), 0, stream>>>(
            vstate, t_seq, wsB, t);
        k_edge_gemm<<<dim3(NMAT, NJT / 8), dim3(256), 0, stream>>>(
            wsA, wsB, sstate, sinstate, src);
        k_update<<<dim3(NP * NB * NN / 256), dim3(256), 0, stream>>>(
            vstate, sstate, sinstate, tgt, (float*)d_out, errPart, t);
    }
    k_err<<<dim3(1), dim3(256), 0, stream>>>(
        errPart, (float*)d_out + (size_t)NB * NT * NP * NN);
}
